// AntiviralGNN_30872224924220
// MI455X (gfx1250) — matrix operand order repaired and checked by reading, not compiled, not run on hardware
//
#include <hip/hip_runtime.h>
#include <hip/hip_bf16.h>

typedef __attribute__((ext_vector_type(16))) __bf16 v16bf;
typedef __attribute__((ext_vector_type(8)))  float  v8f;
typedef __attribute__((ext_vector_type(4)))  unsigned int v4u;

constexpr int kN  = 131072;
constexpr int kE  = 262144;
constexpr int kG  = 4096;
constexpr int kFN = 30;
constexpr int kFE = 11;
constexpr int kH  = 256;
constexpr int kH2 = 512;
constexpr int kL  = 5;
constexpr float kBnEps = 1e-5f;

// ---------- helpers ----------

__device__ __forceinline__ unsigned short f2bf(float f) {
  unsigned int u = __float_as_uint(f);
  u = (u + 0x7FFFu + ((u >> 16) & 1u)) >> 16;   // round-nearest-even
  return (unsigned short)u;
}

__device__ __forceinline__ void atomicMaxF(float* addr, float val) {
  if (val >= 0.0f) atomicMax((int*)addr, __float_as_int(val));
  else             atomicMin((unsigned int*)addr, __float_as_uint(val));
}

// Fragment load: 8 consecutive packed uints per lane -> 2x ds_load_b128
__device__ __forceinline__ v16bf frag_from_lds(const unsigned int* p) {
  union { v4u q[2]; v16bf v; } f;
  f.q[0] = *(const v4u*)(p);
  f.q[1] = *(const v4u*)(p + 4);
  return f.v;
}

// Map k (0..31, even) within a 32-slab to the wmma A/B lane-half and j slot.
//   lanes 0-15 hold K 0..7 & 16..23, lanes 16-31 hold K 8..15 & 24..31 (pairs).
__device__ __forceinline__ int k_to_lanehi(int k) { return (k >> 3) & 1; }
__device__ __forceinline__ int k_to_j(int k)      { return ((k & 7) >> 1) + ((k >> 4) << 2); }

// Generic (flat) shared pointer -> LDS byte offset (low 32 bits of the address)
__device__ __forceinline__ unsigned int lds_off(const void* p) {
  return (unsigned int)(unsigned long long)p;
}

// Async global->LDS copies (CDNA5, ASYNCcnt-tracked); bypasses VGPR round trip.
__device__ __forceinline__ void async_copy_b128(unsigned int ldsDst, const void* gsrc) {
  asm volatile("global_load_async_to_lds_b128 %0, %1, off"
               :: "v"(ldsDst), "v"((unsigned long long)gsrc) : "memory");
}
__device__ __forceinline__ void async_copy_b32(unsigned int ldsDst, const void* gsrc) {
  asm volatile("global_load_async_to_lds_b32 %0, %1, off"
               :: "v"(ldsDst), "v"((unsigned long long)gsrc) : "memory");
}
__device__ __forceinline__ void wait_async() {
  asm volatile("s_wait_asynccnt 0" ::: "memory");
}

// ---------- elementwise utility kernels ----------

__global__ void k_fill(float* p, float v, long long n) {
  long long i = (long long)blockIdx.x * 256 + threadIdx.x;
  long long s = (long long)gridDim.x * 256;
  for (; i < n; i += s) p[i] = v;
}

// Swizzle f32 weights [L][K][Ncols] into fragment-ready packed bf16 uints:
//   out[((l*(K/32)+slab)*(Ncols/16)+ct)*256 + lane*8 + j] = pack(bf16 pair along K)
__global__ void k_swizzle_w(const float* __restrict__ w, unsigned int* __restrict__ out,
                            int K, int Ncols, int layers) {
  long long total = (long long)layers * (K / 32) * (Ncols / 16) * 256;
  int nt = Ncols / 16, ns = K / 32;
  long long u = (long long)blockIdx.x * 256 + threadIdx.x;
  long long st = (long long)gridDim.x * 256;
  for (; u < total; u += st) {
    int j    = (int)(u & 7);
    int lane = (int)((u >> 3) & 31);
    long long rest = u >> 8;
    int ct = (int)(rest % nt);
    long long r2 = rest / nt;
    int s = (int)(r2 % ns);
    int l = (int)(r2 / ns);
    int n  = lane & 15;
    int k  = s * 32 + (lane >> 4) * 8 + (j >> 2) * 16 + 2 * (j & 3);
    long long base = (long long)l * K * Ncols + (long long)k * Ncols + ct * 16 + n;
    unsigned int lo = f2bf(w[base]);
    unsigned int hi = f2bf(w[base + Ncols]);
    out[u] = lo | (hi << 16);
  }
}

// ---------- node encoder: h = x @ enc_w + enc_b ----------

__global__ __launch_bounds__(256) void k_encode_nodes(
    const float* __restrict__ x, const float* __restrict__ w,
    const float* __restrict__ b, float* __restrict__ h) {
  __shared__ float xr[kFN];
  int node = blockIdx.x, c = threadIdx.x;
  if (c < kFN) xr[c] = x[node * kFN + c];
  __syncthreads();
  float a = b[c];
#pragma unroll
  for (int k = 0; k < kFN; ++k) a = fmaf(xr[k], w[k * kH + c], a);
  h[node * kH + c] = a;
}

// ---------- message + scatter: agg[dst] += relu(h[src] + (ea@bond_w + bond_b)) ----------

__global__ __launch_bounds__(256) void k_message(
    const float* __restrict__ h, const int* __restrict__ ei, const float* __restrict__ ea,
    const float* __restrict__ bw, const float* __restrict__ bb, float* __restrict__ agg) {
  __shared__ float bws[kFE * kH];
  __shared__ float eas[16][kFE];
  __shared__ int se[16], de[16];
  int tid = threadIdx.x;
  int e0 = blockIdx.x * 16;
  for (int i = tid; i < kFE * kH; i += 256) bws[i] = bw[i];
  if (tid < 16 * kFE) { int e = tid / kFE, k = tid % kFE; eas[e][k] = ea[(e0 + e) * kFE + k]; }
  if (tid < 16) { se[tid] = ei[e0 + tid]; de[tid] = ei[kE + e0 + tid]; }
  __syncthreads();
  int c = tid;
  float bbc = bb[c];
  for (int e = 0; e < 16; ++e) {
    float acc = bbc;
#pragma unroll
    for (int k = 0; k < kFE; ++k) acc = fmaf(eas[e][k], bws[k * kH + c], acc);
    float m = h[(long long)se[e] * kH + c] + acc;
    m = m > 0.0f ? m : 0.0f;
    atomicAdd(&agg[(long long)de[e] * kH + c], m);
  }
}

// ---------- GEMM1: t = relu(BN( ((1+eps)h + agg) @ w1 + b1 ))  -> bf16 [N, 512] ----------

__global__ __launch_bounds__(256) void k_gemm1(
    const float* __restrict__ h, const float* __restrict__ agg,
    const float* __restrict__ eps, int layer,
    const unsigned int* __restrict__ wswz, const float* __restrict__ b1,
    const float* __restrict__ bng, const float* __restrict__ bnb,
    const float* __restrict__ bnm, const float* __restrict__ bnv,
    unsigned short* __restrict__ tout) {
  __shared__ unsigned int Asw[2][8 * 32 * 8];   // double-buffered A fragments (16KB)
  __shared__ unsigned int Bsw[2][4 * 32 * 8];   // double-buffered B fragments (8KB)
  const int tid = threadIdx.x, lane = tid & 31, wave = tid >> 5;
  const int rowBase = blockIdx.x * 128, colBase = blockIdx.y * 64;
  const int numTiles = kH2 / 16, colTile = colBase >> 4;
  const float ep = 1.0f + eps[layer];
  constexpr int S = kH / 32;                    // K-slabs
  v8f acc[4];
  const v8f vz = {0.f, 0.f, 0.f, 0.f, 0.f, 0.f, 0.f, 0.f};
#pragma unroll
  for (int nt = 0; nt < 4; ++nt) acc[nt] = vz;

  auto stage = [&](int slab, int buf) {
    int k0 = slab * 32;
    // A: fused z formation + bf16 convert, stored in fragment layout (VALU path)
#pragma unroll
    for (int i = 0; i < 8; ++i) {
      int idx = i * 256 + tid;
      int r = idx >> 4, k = (idx & 15) * 2;
      int gi = (rowBase + r) * kH + k0 + k;
      float2 hv = *(const float2*)(h + gi);
      float2 av = *(const float2*)(agg + gi);
      unsigned int pv = (unsigned int)f2bf(ep * hv.x + av.x) |
                        ((unsigned int)f2bf(ep * hv.y + av.y) << 16);
      Asw[buf][(((r >> 4) * 32) + (r & 15) + (k_to_lanehi(k) << 4)) * 8 + k_to_j(k)] = pv;
    }
    // B: pre-swizzled weights, straight async copy global->LDS (no VGPR round trip)
    const unsigned int* wsl = wswz + ((size_t)slab * numTiles + colTile) * 256;
    async_copy_b128(lds_off(&Bsw[buf][tid * 4]), wsl + tid * 4);
  };

  stage(0, 0);
  wait_async();
  __syncthreads();

  for (int slab = 0; slab < S; ++slab) {
    int cur = slab & 1;
    if (slab + 1 < S) stage(slab + 1, cur ^ 1);
    v16bf a = frag_from_lds(&Asw[cur][(wave * 32 + lane) * 8]);
#pragma unroll
    for (int nt = 0; nt < 4; ++nt) {
      v16bf bfr = frag_from_lds(&Bsw[cur][(nt * 32 + lane) * 8]);
      acc[nt] = __builtin_amdgcn_wmma_f32_16x16x32_bf16(
          false, a, false, bfr, (short)0, acc[nt], false, false);
    }
    wait_async();
    __syncthreads();
  }

  const int ncol  = lane & 15;
  const int rbase = wave * 16 + ((lane >> 4) << 3);
#pragma unroll
  for (int nt = 0; nt < 4; ++nt) {
    int col = colBase + nt * 16 + ncol;
    float gg = bng[col], bb = bnb[col], mm = bnm[col];
    float iv = rsqrtf(bnv[col] + kBnEps);
    float bc = b1[col];
#pragma unroll
    for (int i = 0; i < 8; ++i) {
      int row = rowBase + rbase + i;
      float v = acc[nt][i] + bc;
      v = gg * (v - mm) * iv + bb;
      v = v > 0.0f ? v : 0.0f;
      tout[(long long)row * kH2 + col] = f2bf(v);
    }
  }
}

// ---------- GEMM2: h = relu(BN( t @ w2 + b2 )) + h  (in place, f32) ----------

__global__ __launch_bounds__(256) void k_gemm2(
    const unsigned short* __restrict__ tin,
    const unsigned int* __restrict__ wswz, const float* __restrict__ b2,
    const float* __restrict__ bng, const float* __restrict__ bnb,
    const float* __restrict__ bnm, const float* __restrict__ bnv,
    float* __restrict__ h) {
  __shared__ unsigned int Asw[2][8 * 32 * 8];
  __shared__ unsigned int Bsw[2][4 * 32 * 8];
  const int tid = threadIdx.x, lane = tid & 31, wave = tid >> 5;
  const int rowBase = blockIdx.x * 128, colBase = blockIdx.y * 64;
  const int numTiles = kH / 16, colTile = colBase >> 4;
  constexpr int S = kH2 / 32;
  v8f acc[4];
  const v8f vz = {0.f, 0.f, 0.f, 0.f, 0.f, 0.f, 0.f, 0.f};
#pragma unroll
  for (int nt = 0; nt < 4; ++nt) acc[nt] = vz;

  auto stage = [&](int slab, int buf) {
    int k0 = slab * 32;
    // A: bf16 pairs already packed in t; async b32 scatter does the swizzle in-flight
#pragma unroll
    for (int i = 0; i < 8; ++i) {
      int idx = i * 256 + tid;
      int r = idx >> 4, k = (idx & 15) * 2;
      long long gi = (long long)(rowBase + r) * kH2 + k0 + k;
      unsigned int dst = lds_off(
          &Asw[buf][(((r >> 4) * 32) + (r & 15) + (k_to_lanehi(k) << 4)) * 8 + k_to_j(k)]);
      async_copy_b32(dst, tin + gi);
    }
    // B: contiguous async copy of pre-swizzled weights
    const unsigned int* wsl = wswz + ((size_t)slab * numTiles + colTile) * 256;
    async_copy_b128(lds_off(&Bsw[buf][tid * 4]), wsl + tid * 4);
  };

  stage(0, 0);
  wait_async();
  __syncthreads();

  for (int slab = 0; slab < S; ++slab) {
    int cur = slab & 1;
    if (slab + 1 < S) stage(slab + 1, cur ^ 1);
    v16bf a = frag_from_lds(&Asw[cur][(wave * 32 + lane) * 8]);
#pragma unroll
    for (int nt = 0; nt < 4; ++nt) {
      v16bf bfr = frag_from_lds(&Bsw[cur][(nt * 32 + lane) * 8]);
      acc[nt] = __builtin_amdgcn_wmma_f32_16x16x32_bf16(
          false, a, false, bfr, (short)0, acc[nt], false, false);
    }
    wait_async();
    __syncthreads();
  }

  const int ncol  = lane & 15;
  const int rbase = wave * 16 + ((lane >> 4) << 3);
#pragma unroll
  for (int nt = 0; nt < 4; ++nt) {
    int col = colBase + nt * 16 + ncol;
    float gg = bng[col], bb = bnb[col], mm = bnm[col];
    float iv = rsqrtf(bnv[col] + kBnEps);
    float bc = b2[col];
#pragma unroll
    for (int i = 0; i < 8; ++i) {
      int row = rowBase + rbase + i;
      float v = acc[nt][i] + bc;
      v = gg * (v - mm) * iv + bb;
      v = v > 0.0f ? v : 0.0f;
      long long o = (long long)row * kH + col;
      h[o] = v + h[o];                     // residual, read-then-write same element
    }
  }
}

// ---------- pooling ----------

__global__ void k_counts(const int* __restrict__ batch, float* __restrict__ counts) {
  int i = blockIdx.x * 256 + threadIdx.x;
  int s = gridDim.x * 256;
  for (; i < kN; i += s) atomicAdd(&counts[batch[i]], 1.0f);
}

__global__ void k_pool(const float* __restrict__ h, const int* __restrict__ batch,
                       float* __restrict__ sums, float* __restrict__ maxs) {
  long long total = (long long)kN * kH;
  long long i = (long long)blockIdx.x * 256 + threadIdx.x;
  long long s = (long long)gridDim.x * 256;
  for (; i < total; i += s) {
    int node = (int)(i >> 8), c = (int)(i & 255);
    int g = batch[node];
    float v = h[i];
    atomicAdd(&sums[g * kH + c], v);
    atomicMaxF(&maxs[g * kH + c], v);
  }
}

// ---------- classifier: one block per graph ----------

__global__ __launch_bounds__(256) void k_classifier(
    const float* __restrict__ sums, const float* __restrict__ maxs, const float* __restrict__ counts,
    const float* __restrict__ cw1, const float* __restrict__ cb1,
    const float* __restrict__ c1g, const float* __restrict__ c1b,
    const float* __restrict__ c1m, const float* __restrict__ c1v,
    const float* __restrict__ cw2, const float* __restrict__ cb2,
    const float* __restrict__ c2g, const float* __restrict__ c2b,
    const float* __restrict__ c2m, const float* __restrict__ c2v,
    const float* __restrict__ cw3, const float* __restrict__ cb3,
    float* __restrict__ out) {
  __shared__ float p[kH2];
  __shared__ float q1[kH];
  __shared__ float red[kH / 2];
  int g = blockIdx.x, tid = threadIdx.x;
  float cnt = counts[g];
  p[tid] = sums[g * kH + tid] / fmaxf(cnt, 1.0f);
  float mx = maxs[g * kH + tid];
  p[kH + tid] = (cnt > 0.0f) ? mx : 0.0f;
  __syncthreads();

  float a = cb1[tid];
  for (int k = 0; k < kH2; ++k) a = fmaf(p[k], cw1[k * kH + tid], a);
  a = c1g[tid] * (a - c1m[tid]) * rsqrtf(c1v[tid] + kBnEps) + c1b[tid];
  q1[tid] = a > 0.0f ? a : 0.0f;
  __syncthreads();

  if (tid < kH / 2) {
    float a2 = cb2[tid];
    for (int k = 0; k < kH; ++k) a2 = fmaf(q1[k], cw2[k * (kH / 2) + tid], a2);
    a2 = c2g[tid] * (a2 - c2m[tid]) * rsqrtf(c2v[tid] + kBnEps) + c2b[tid];
    a2 = a2 > 0.0f ? a2 : 0.0f;
    red[tid] = a2 * cw3[tid];
  }
  __syncthreads();
  for (int s = kH / 4; s > 0; s >>= 1) {
    if (tid < s) red[tid] += red[tid + s];
    __syncthreads();
  }
  if (tid == 0) out[g] = red[0] + cb3[0];
}

// ---------- launch ----------

extern "C" void kernel_launch(void* const* d_in, const int* in_sizes, int n_in,
                              void* d_out, int out_size, void* d_ws, size_t ws_size,
                              hipStream_t stream) {
  const float* x      = (const float*)d_in[0];
  const int*   ei     = (const int*)d_in[1];
  const int*   batch  = (const int*)d_in[2];
  const float* ea     = (const float*)d_in[3];
  const float* enc_w  = (const float*)d_in[4];
  const float* enc_b  = (const float*)d_in[5];
  const float* bond_w = (const float*)d_in[6];
  const float* bond_b = (const float*)d_in[7];
  const float* eps    = (const float*)d_in[8];
  const float* w1     = (const float*)d_in[9];
  const float* b1     = (const float*)d_in[10];
  const float* ibn_g  = (const float*)d_in[11];
  const float* ibn_b  = (const float*)d_in[12];
  const float* ibn_m  = (const float*)d_in[13];
  const float* ibn_v  = (const float*)d_in[14];
  const float* w2     = (const float*)d_in[15];
  const float* b2     = (const float*)d_in[16];
  const float* obn_g  = (const float*)d_in[17];
  const float* obn_b  = (const float*)d_in[18];
  const float* obn_m  = (const float*)d_in[19];
  const float* obn_v  = (const float*)d_in[20];
  const float* cw1    = (const float*)d_in[21];
  const float* cb1    = (const float*)d_in[22];
  const float* c1g    = (const float*)d_in[23];
  const float* c1b    = (const float*)d_in[24];
  const float* c1m    = (const float*)d_in[25];
  const float* c1v    = (const float*)d_in[26];
  const float* cw2    = (const float*)d_in[27];
  const float* cb2    = (const float*)d_in[28];
  const float* c2g    = (const float*)d_in[29];
  const float* c2b    = (const float*)d_in[30];
  const float* c2m    = (const float*)d_in[31];
  const float* c2v    = (const float*)d_in[32];
  const float* cw3    = (const float*)d_in[33];
  const float* cb3    = (const float*)d_in[34];
  float* out = (float*)d_out;

  char* ws = (char*)d_ws;
  float*          h_buf  = (float*)(ws);                                  // N*H f32
  float*          agg    = (float*)(ws + 134217728ULL);                   // N*H f32
  unsigned short* t_buf  = (unsigned short*)(ws + 268435456ULL);          // N*2H bf16
  unsigned int*   w1swz  = (unsigned int*)(ws + 402653184ULL);            // L*H*2H/2 uints
  unsigned int*   w2swz  = (unsigned int*)(ws + 405274624ULL);            // L*2H*H/2 uints
  float*          sums   = (float*)(ws + 407896064ULL);                   // G*H
  float*          maxs   = (float*)(ws + 412090368ULL);                   // G*H
  float*          counts = (float*)(ws + 416284672ULL);                   // G

  k_swizzle_w<<<512, 256, 0, stream>>>(w1, w1swz, kH, kH2, kL);
  k_swizzle_w<<<512, 256, 0, stream>>>(w2, w2swz, kH2, kH, kL);
  k_encode_nodes<<<kN, 256, 0, stream>>>(x, enc_w, enc_b, h_buf);

  for (int l = 0; l < kL; ++l) {
    k_fill<<<4096, 256, 0, stream>>>(agg, 0.0f, (long long)kN * kH);
    k_message<<<kE / 16, 256, 0, stream>>>(h_buf, ei, ea, bond_w, bond_b, agg);
    dim3 g1(kN / 128, kH2 / 64);
    k_gemm1<<<g1, 256, 0, stream>>>(h_buf, agg, eps, l,
        w1swz + (size_t)l * (kH * kH2 / 2), b1 + l * kH2,
        ibn_g + l * kH2, ibn_b + l * kH2, ibn_m + l * kH2, ibn_v + l * kH2, t_buf);
    dim3 g2(kN / 128, kH / 64);
    k_gemm2<<<g2, 256, 0, stream>>>(t_buf,
        w2swz + (size_t)l * (kH2 * kH / 2), b2 + l * kH,
        obn_g + l * kH, obn_b + l * kH, obn_m + l * kH, obn_v + l * kH, h_buf);
  }

  k_fill<<<2048, 256, 0, stream>>>(sums, 0.0f, (long long)kG * kH);
  k_fill<<<2048, 256, 0, stream>>>(maxs, -__builtin_inff(), (long long)kG * kH);
  k_fill<<<64, 256, 0, stream>>>(counts, 0.0f, (long long)kG);
  k_counts<<<512, 256, 0, stream>>>(batch, counts);
  k_pool<<<4096, 256, 0, stream>>>(h_buf, batch, sums, maxs);
  k_classifier<<<kG, 256, 0, stream>>>(sums, maxs, counts,
      cw1, cb1, c1g, c1b, c1m, c1v, cw2, cb2, c2g, c2b, c2m, c2v, cw3, cb3, out);
}
